// GINConvNet_17927193493730
// MI455X (gfx1250) — matrix operand order repaired and checked by reading, not compiled, not run on hardware
//
#include <hip/hip_runtime.h>
#include <hip/hip_bf16.h>

#define N_NODES 400000
#define N_EDGES 1600000
#define BATCH   4096
#define DIN0    78
#define DIM     32

typedef __attribute__((ext_vector_type(16))) _Float16 v16h;
typedef __attribute__((ext_vector_type(8)))  _Float16 v8h;
typedef __attribute__((ext_vector_type(8)))  float    v8f;

__device__ __forceinline__ int lane_id(){ return (int)(threadIdx.x & 31u); }
__device__ __forceinline__ int wave_id(){ return (int)(threadIdx.x >> 5u); }

// A-matrix (16x32 f16) K index for VGPR pair j, lane-half h (ISA 7.12.2)
__device__ __forceinline__ int a_kb(int j, int h){ return (j < 4) ? (h*8 + 2*j) : (16 + h*8 + 2*(j-4)); }

__device__ __forceinline__ v8f zero8(){
  v8f z;
#pragma unroll
  for (int i = 0; i < 8; ++i) z[i] = 0.f;
  return z;
}

__device__ __forceinline__ v8f wmma32(v16h a, v16h b, v8f c){
  // D = A(16x32 f16) x B(32x16 f16) + C(f32)
  return __builtin_amdgcn_wmma_f32_16x16x32_f16(false, a, false, b, (short)0, c, false, false);
}

// ---- fragment loaders -------------------------------------------------------

// A fragment from contiguous f16 row: two aligned 16B vector loads
__device__ __forceinline__ v16h load_a_rowf16(const _Float16* rowbase, int k0){
  int h = lane_id() >> 4;
  const v8h* p0 = (const v8h*)(rowbase + k0 + 8 * h);
  const v8h* p1 = (const v8h*)(rowbase + k0 + 16 + 8 * h);
  v8h x = *p0, y = *p1;
  return __builtin_shufflevector(x, y, 0,1,2,3,4,5,6,7,8,9,10,11,12,13,14,15);
}

// A fragment from contiguous f32 row (no clamp, unconditional)
__device__ __forceinline__ v16h load_a_f32(const float* rowbase, int k0){
  int h = lane_id() >> 4;
  v16h a;
#pragma unroll
  for (int j = 0; j < 8; ++j){
    int k = k0 + a_kb(j, h);
    a[2*j]   = (_Float16)rowbase[k];
    a[2*j+1] = (_Float16)rowbase[k + 1];
  }
  return a;
}

// B fragment from pre-packed f16 weights: per lane = 16 consecutive halves
// rpblk points at the (kc,nt) block of 32 lanes x 16 halves (512 halves)
__device__ __forceinline__ v16h load_b_rp(const _Float16* rpblk){
  const v8h* p = (const v8h*)(rpblk + (size_t)lane_id() * 16);
  v8h x = p[0], y = p[1];
  return __builtin_shufflevector(x, y, 0,1,2,3,4,5,6,7,8,9,10,11,12,13,14,15);
}

// ---- weight repack kernels --------------------------------------------------
// rp[((kc*NT+nt)*32 + lane)*16 + e] = (f16) W[k*ldw + nt*16 + (lane&15)]
// with k = kc*32 + 8*(lane>>4) + e + (e>=8 ? 8 : 0)  (zero-padded past Kmax)
__global__ void k_repack_w(const float* __restrict__ W, _Float16* __restrict__ rp,
                           int ldw, int NT, int KC, int Kmax){
  long long i = (long long)blockIdx.x * blockDim.x + threadIdx.x;
  long long total = (long long)KC * NT * 512;
  if (i >= total) return;
  int e    = (int)(i & 15);
  int lane = (int)((i >> 4) & 31);
  long long t = i >> 9;
  int nt = (int)(t % NT);
  int kc = (int)(t / NT);
  int k   = kc * 32 + 8 * (lane >> 4) + e + (e >= 8 ? 8 : 0);
  int col = nt * 16 + (lane & 15);
  float v = (k < Kmax) ? W[(size_t)k * ldw + col] : 0.f;
  rp[i] = (_Float16)v;
}

// conv weights W[OC][IC][8], implicit im2col K = dk*IC + ic (IC = 2^icShift)
__global__ void k_repack_convw(const float* __restrict__ W, _Float16* __restrict__ rp,
                               int icShift, int NT, int KC){
  long long i = (long long)blockIdx.x * blockDim.x + threadIdx.x;
  long long total = (long long)KC * NT * 512;
  if (i >= total) return;
  int e    = (int)(i & 15);
  int lane = (int)((i >> 4) & 31);
  long long t = i >> 9;
  int nt = (int)(t % NT);
  int kc = (int)(t / NT);
  int k  = kc * 32 + 8 * (lane >> 4) + e + (e >= 8 ? 8 : 0);
  int oc = nt * 16 + (lane & 15);
  int IC = 1 << icShift;
  int ic = k & (IC - 1), dk = k >> icShift;
  rp[i] = (_Float16)W[((size_t)oc * IC + ic) * 8 + dk];
}

// ---- utility kernels --------------------------------------------------------

__global__ void k_zero(float* __restrict__ p, long long n){
  long long i = (long long)blockIdx.x * blockDim.x + threadIdx.x;
  if (i < n) p[i] = 0.f;
}

// edge scatter: agg[dst] += h[src], lane = channel (wave32-coalesced, atomics hit L2)
__global__ void k_edge_agg(const float* __restrict__ h, const int* __restrict__ src,
                           const int* __restrict__ dst, float* __restrict__ agg,
                           int D, int Dp, long long total){
  long long i = (long long)blockIdx.x * blockDim.x + threadIdx.x;
  if (i >= total) return;
  int c = (int)(i % Dp);
  long long e = i / Dp;
  if (c >= D) return;
  int s = src[e], d = dst[e];
  atomicAdd(&agg[(long long)d * D + c], h[(long long)s * D + c]);
}

// fused GIN MLP: t = relu(relu((h+agg)@W1+b1)@W2+b2); BN partial sums via atomics
template<int DIN>
__global__ __launch_bounds__(256)
void k_gin_mlp(const float* __restrict__ hin, const float* __restrict__ agg,
               const _Float16* __restrict__ rpW1, const float* __restrict__ b1,
               const _Float16* __restrict__ rpW2, const float* __restrict__ b2,
               float* __restrict__ tout, float* __restrict__ gsum, float* __restrict__ gsq){
  __shared__ _Float16 lds[8 * 16 * 32];         // 8KB: one 16x32 f16 tile per wave
  constexpr int KCH = (DIN + 31) / 32;
  int w = wave_id(), l = lane_id();
  int row0 = (blockIdx.x * 8 + w) * 16;
  int r = l & 15, hh = l >> 4, c = l & 15;

  const float* hrow = hin + (long long)(row0 + r) * DIN;
  const float* arow = agg + (long long)(row0 + r) * DIN;

  v8f acc0 = zero8(), acc1 = zero8();
#pragma unroll
  for (int kc = 0; kc < KCH; ++kc){
    const int k0 = kc * 32;
    v16h a;
#pragma unroll
    for (int j = 0; j < 8; ++j){
      int k = k0 + a_kb(j, hh);
      float x0, x1;
      if (k0 + 32 <= DIN){                      // full chunk: unconditional vector loads
        x0 = hrow[k] + arow[k];
        x1 = hrow[k + 1] + arow[k + 1];
      } else {                                  // partial chunk: clamped addr + cndmask
        int ka = (k     < DIN) ? k     : (DIN - 1);
        int kb = (k + 1 < DIN) ? k + 1 : (DIN - 1);
        float t0 = hrow[ka] + arow[ka];
        float t1 = hrow[kb] + arow[kb];
        x0 = (k     < DIN) ? t0 : 0.f;
        x1 = (k + 1 < DIN) ? t1 : 0.f;
      }
      a[2*j] = (_Float16)x0; a[2*j+1] = (_Float16)x1;
    }
    acc0 = wmma32(a, load_b_rp(rpW1 + (size_t)(kc * 2 + 0) * 512), acc0);
    acc1 = wmma32(a, load_b_rp(rpW1 + (size_t)(kc * 2 + 1) * 512), acc1);
  }

  // bias + relu -> LDS f16 (re-fragment C layout -> A layout)
  _Float16* slab = &lds[w * 512];
#pragma unroll
  for (int rr = 0; rr < 8; ++rr){
    int m = rr + 8 * hh;
    float v0 = acc0[rr] + b1[c];       slab[m * 32 + c]      = (_Float16)(v0 > 0.f ? v0 : 0.f);
    float v1 = acc1[rr] + b1[16 + c];  slab[m * 32 + 16 + c] = (_Float16)(v1 > 0.f ? v1 : 0.f);
  }
  __syncthreads();

  v16h a2 = load_a_rowf16(slab + r * 32, 0);
  v8f d0 = zero8(), d1 = zero8();
  d0 = wmma32(a2, load_b_rp(rpW2), d0);
  d1 = wmma32(a2, load_b_rp(rpW2 + 512), d1);

  float s0 = 0.f, q0 = 0.f, s1 = 0.f, q1 = 0.f;
#pragma unroll
  for (int rr = 0; rr < 8; ++rr){
    int m = rr + 8 * hh;
    float v0 = d0[rr] + b2[c];      v0 = v0 > 0.f ? v0 : 0.f;
    float v1 = d1[rr] + b2[16 + c]; v1 = v1 > 0.f ? v1 : 0.f;
    tout[(long long)(row0 + m) * 32 + c]      = v0;
    tout[(long long)(row0 + m) * 32 + 16 + c] = v1;
    s0 += v0; q0 += v0 * v0; s1 += v1; q1 += v1 * v1;
  }
  atomicAdd(&gsum[c], s0);      atomicAdd(&gsq[c], q0);
  atomicAdd(&gsum[16 + c], s1); atomicAdd(&gsq[16 + c], q1);
}

// finalize BN stats and apply
__global__ void k_bn_apply(const float* __restrict__ t, const float* __restrict__ gsum,
                           const float* __restrict__ gsq, const float* __restrict__ gam,
                           const float* __restrict__ bet, float* __restrict__ hout, int Nrows){
  long long i = (long long)blockIdx.x * blockDim.x + threadIdx.x;
  long long total = (long long)Nrows * 32;
  if (i >= total) return;
  int c = (int)(i & 31);
  float inv = 1.f / (float)Nrows;
  float mu  = gsum[c] * inv;
  float var = gsq[c] * inv - mu * mu;
  hout[i] = gam[c] * (t[i] - mu) * rsqrtf(var + 1e-5f) + bet[c];
}

// graph pooling: g[batch[n]] += h[n]
__global__ void k_pool(const float* __restrict__ h, const int* __restrict__ batch,
                       float* __restrict__ g, int Nrows){
  long long i = (long long)blockIdx.x * blockDim.x + threadIdx.x;
  if (i >= (long long)Nrows * 32) return;
  int c = (int)(i & 31);
  long long n = i >> 5;
  atomicAdd(&g[(long long)batch[n] * 32 + c], h[i]);
}

// xd = relu(g @ W(32x128) + b), dual store: f32 to d_out, f16 to concat buffer
__global__ __launch_bounds__(256)
void k_xd(const float* __restrict__ g, const _Float16* __restrict__ rpW,
          const float* __restrict__ b, float* __restrict__ xd, _Float16* __restrict__ xc){
  int w = wave_id(), l = lane_id();
  int row0 = (blockIdx.x * 8 + w) * 16;
  int r = l & 15, hh = l >> 4, c = l & 15;
  v16h a = load_a_f32(g + (long long)(row0 + r) * 32, 0);
#pragma unroll
  for (int t = 0; t < 8; ++t){
    v8f acc = zero8();
    acc = wmma32(a, load_b_rp(rpW + (size_t)t * 512), acc);
#pragma unroll
    for (int rr = 0; rr < 8; ++rr){
      int m = rr + 8 * hh, cc = t * 16 + c;
      float v = acc[rr] + b[cc]; v = v > 0.f ? v : 0.f;
      xd[(long long)(row0 + m) * 128 + cc] = v;
      xc[(long long)(row0 + m) * 256 + cc] = (_Float16)v;
    }
  }
}

// conv1 (IC=1,K=8) + relu + maxpool3, channels-last f16 out: (B,311,32)
__global__ void k_conv1(const float* __restrict__ tgt, const float* __restrict__ W,
                        const float* __restrict__ b, _Float16* __restrict__ out){
  long long i = (long long)blockIdx.x * blockDim.x + threadIdx.x;
  if (i >= 4096LL * 311 * 32) return;
  int oc = (int)(i & 31);
  long long j = i >> 5;
  int p  = (int)(j % 311);
  int bb = (int)(j / 311);
  const float* trow = tgt + (long long)bb * 942;
  const float* wk = W + oc * 8;
  float bias = b[oc];
  float m = 0.f;
#pragma unroll
  for (int q = 0; q < 3; ++q){
    int pos = p * 3 + q;
    float s = bias;
#pragma unroll
    for (int dk = 0; dk < 8; ++dk) s += trow[pos + dk] * wk[dk];
    s = s > 0.f ? s : 0.f;
    m = s > m ? s : m;
  }
  out[i] = (_Float16)m;
}

// conv2 as implicit-im2col WMMA GEMM: (B*304) x 64, K=256; relu; out (B,304,64) f16
__global__ __launch_bounds__(256)
void k_conv2(const _Float16* __restrict__ in, const _Float16* __restrict__ rpW,
             const float* __restrict__ b, _Float16* __restrict__ out){
  int w = wave_id(), l = lane_id();
  long long row0 = ((long long)blockIdx.x * 8 + w) * 16;
  int r = l & 15, hh = l >> 4, c = l & 15;
  long long gr = row0 + r;
  int p = (int)(gr % 304), bb = (int)(gr / 304);
  const _Float16* rowb = in + ((long long)bb * 311 + p) * 32;   // contiguous K=dk*32+ic

  v8f acc[4]; for (int t = 0; t < 4; ++t) acc[t] = zero8();
#pragma unroll
  for (int kc = 0; kc < 8; ++kc){
    v16h a = load_a_rowf16(rowb, kc * 32);
#pragma unroll
    for (int t = 0; t < 4; ++t)
      acc[t] = wmma32(a, load_b_rp(rpW + (size_t)(kc * 4 + t) * 512), acc[t]);
  }
#pragma unroll
  for (int t = 0; t < 4; ++t){
#pragma unroll
    for (int rr = 0; rr < 8; ++rr){
      long long gm = row0 + rr + 8 * hh;
      int pp = (int)(gm % 304), bo = (int)(gm / 304);
      int oc = t * 16 + c;
      float v = acc[t][rr] + b[oc]; v = v > 0.f ? v : 0.f;
      out[((long long)bo * 304 + pp) * 64 + oc] = (_Float16)v;
    }
  }
}

// maxpool3 channels-last f16
__global__ void k_pool3_cl(const _Float16* __restrict__ in, _Float16* __restrict__ out,
                           int Lin, int Lout, int C, long long total){
  long long i = (long long)blockIdx.x * blockDim.x + threadIdx.x;
  if (i >= total) return;
  int c = (int)(i % C);
  long long j = i / C;
  int p  = (int)(j % Lout);
  int bb = (int)(j / Lout);
  const _Float16* base = in + ((long long)bb * Lin + p * 3) * C + c;
  float m = fmaxf((float)base[0], fmaxf((float)base[C], (float)base[2 * C]));
  out[i] = (_Float16)m;
}

// conv3 WMMA GEMM: (B*94) x 128, K=512; relu; out (B,94,128) f16
__global__ __launch_bounds__(256)
void k_conv3(const _Float16* __restrict__ in, const _Float16* __restrict__ rpW,
             const float* __restrict__ b, _Float16* __restrict__ out){
  int w = wave_id(), l = lane_id();
  long long row0 = ((long long)blockIdx.x * 8 + w) * 16;
  int r = l & 15, hh = l >> 4, c = l & 15;
  long long gr = row0 + r;
  int p = (int)(gr % 94), bb = (int)(gr / 94);
  const _Float16* rowb = in + ((long long)bb * 101 + p) * 64;   // contiguous K=dk*64+ic

  v8f acc[8]; for (int t = 0; t < 8; ++t) acc[t] = zero8();
#pragma unroll
  for (int kc = 0; kc < 16; ++kc){
    v16h a = load_a_rowf16(rowb, kc * 32);
#pragma unroll
    for (int t = 0; t < 8; ++t)
      acc[t] = wmma32(a, load_b_rp(rpW + (size_t)(kc * 8 + t) * 512), acc[t]);
  }
#pragma unroll
  for (int t = 0; t < 8; ++t){
#pragma unroll
    for (int rr = 0; rr < 8; ++rr){
      long long gm = row0 + rr + 8 * hh;
      int pp = (int)(gm % 94), bo = (int)(gm / 94);
      int oc = t * 16 + c;
      float v = acc[t][rr] + b[oc]; v = v > 0.f ? v : 0.f;
      out[((long long)bo * 94 + pp) * 128 + oc] = (_Float16)v;
    }
  }
}

// maxpool3 + transpose to NCH flat: out[b*3968 + ch*31 + p] (matches reference reshape)
__global__ void k_pool3_tr(const _Float16* __restrict__ in, _Float16* __restrict__ out,
                           long long total){
  long long i = (long long)blockIdx.x * blockDim.x + threadIdx.x;
  if (i >= total) return;
  int p = (int)(i % 31);
  long long j = i / 31;
  int c  = (int)(j % 128);
  int bb = (int)(j / 128);
  const _Float16* base = in + ((long long)bb * 94 + p * 3) * 128 + c;
  float m = fmaxf((float)base[0], fmaxf((float)base[128], (float)base[256]));
  out[(long long)bb * 3968 + c * 31 + p] = (_Float16)m;
}

// xt = cfeat(4096x3968) @ W(3968x128) + b  -> xc[:,128:256] (no relu)
__global__ __launch_bounds__(256)
void k_xt(const _Float16* __restrict__ cf, const _Float16* __restrict__ rpW,
          const float* __restrict__ b, _Float16* __restrict__ xc){
  int w = wave_id(), l = lane_id();
  int row0 = (blockIdx.x * 8 + w) * 16;
  int r = l & 15, hh = l >> 4, c = l & 15;
  const _Float16* rowb = cf + (long long)(row0 + r) * 3968;

  v8f acc[8]; for (int t = 0; t < 8; ++t) acc[t] = zero8();
  for (int kc = 0; kc < 124; ++kc){
    __builtin_prefetch(rowb + (kc + 1) * 32, 0, 0);   // global_prefetch next A chunk
    v16h a = load_a_rowf16(rowb, kc * 32);
#pragma unroll
    for (int t = 0; t < 8; ++t)
      acc[t] = wmma32(a, load_b_rp(rpW + (size_t)(kc * 8 + t) * 512), acc[t]);
  }
#pragma unroll
  for (int t = 0; t < 8; ++t){
#pragma unroll
    for (int rr = 0; rr < 8; ++rr){
      int m = rr + 8 * hh, cc = t * 16 + c;
      xc[(long long)(row0 + m) * 256 + 128 + cc] = (_Float16)(acc[t][rr] + b[cc]);
    }
  }
}

// fc1: relu(xc(4096x256) @ W(256x1024) + b) -> y1 f16
__global__ __launch_bounds__(256)
void k_fc1(const _Float16* __restrict__ xc, const _Float16* __restrict__ rpW,
           const float* __restrict__ b, _Float16* __restrict__ y1){
  int w = wave_id(), l = lane_id();
  int row0 = blockIdx.x * 16;
  int r = l & 15, hh = l >> 4, c = l & 15;
  const _Float16* rowb = xc + (long long)(row0 + r) * 256;

  v8f acc[8]; for (int t = 0; t < 8; ++t) acc[t] = zero8();
#pragma unroll
  for (int kc = 0; kc < 8; ++kc){
    v16h a = load_a_rowf16(rowb, kc * 32);
#pragma unroll
    for (int t = 0; t < 8; ++t){
      int nt = w * 8 + t;
      acc[t] = wmma32(a, load_b_rp(rpW + (size_t)(kc * 64 + nt) * 512), acc[t]);
    }
  }
#pragma unroll
  for (int t = 0; t < 8; ++t){
    int n0 = (w * 8 + t) * 16;
#pragma unroll
    for (int rr = 0; rr < 8; ++rr){
      int m = rr + 8 * hh, cc = n0 + c;
      float v = acc[t][rr] + b[cc]; v = v > 0.f ? v : 0.f;
      y1[(long long)(row0 + m) * 1024 + cc] = (_Float16)v;
    }
  }
}

// fc2: relu(y1(4096x1024) @ W(1024x128) + b) -> y2 f32
__global__ __launch_bounds__(256)
void k_fc2(const _Float16* __restrict__ y1, const _Float16* __restrict__ rpW,
           const float* __restrict__ b, float* __restrict__ y2){
  int w = wave_id(), l = lane_id();
  int row0 = (blockIdx.x * 8 + w) * 16;
  int r = l & 15, hh = l >> 4, c = l & 15;
  const _Float16* rowb = y1 + (long long)(row0 + r) * 1024;

  v8f acc[8]; for (int t = 0; t < 8; ++t) acc[t] = zero8();
#pragma unroll
  for (int kc = 0; kc < 32; ++kc){
    v16h a = load_a_rowf16(rowb, kc * 32);
#pragma unroll
    for (int t = 0; t < 8; ++t)
      acc[t] = wmma32(a, load_b_rp(rpW + (size_t)(kc * 8 + t) * 512), acc[t]);
  }
#pragma unroll
  for (int t = 0; t < 8; ++t){
#pragma unroll
    for (int rr = 0; rr < 8; ++rr){
      int m = rr + 8 * hh, cc = t * 16 + c;
      float v = acc[t][rr] + b[cc]; v = v > 0.f ? v : 0.f;
      y2[(long long)(row0 + m) * 128 + cc] = v;
    }
  }
}

// out = sigmoid(y2 @ W(128x1) + b); one wave per row
__global__ void k_out(const float* __restrict__ y2, const float* __restrict__ W,
                      const float* __restrict__ b, float* __restrict__ out){
  int row = blockIdx.x * 8 + wave_id();
  int l = lane_id();
  float s = 0.f;
  for (int k = l; k < 128; k += 32) s += y2[(long long)row * 128 + k] * W[k];
  for (int off = 16; off > 0; off >>= 1) s += __shfl_down(s, off, 32);
  if (l == 0) out[row] = 1.f / (1.f + expf(-(s + b[0])));
}

// ---- host side --------------------------------------------------------------

// workspace layout (bytes); GNN region dies before conv phase reuses it
static const size_t OFF_AGG = 0;                  // 124,800,000 (N*78*4); reused by cl1
static const size_t OFF_H0  = 124928000;          // 51,200,000
static const size_t OFF_H1  = OFF_H0 + 51200000;
static const size_t OFF_H2  = OFF_H1 + 51200000;  // ends 278,528,000
static const size_t OFF_CL1 = 0;                  // (B,311,32) f16 = 81,526,784
static const size_t OFF_C2P = 81600000;           // (B,304,64) f16 = 159,383,552
static const size_t OFF_CL2 = 241000000;          // (B,101,64) f16 = 52,953,088
static const size_t OFF_C3P = 81600000;           // (B,94,128) f16 = 98,566,144 (reuse C2P)
static const size_t OFF_CF  = 185000000;          // (B,3968)  f16 = 32,505,856
static const size_t OFF_G   = 294000000;          // (B,32) f32 = 524,288
static const size_t OFF_SUM = 294600000;          // gsum(32)+gsq(32) f32
static const size_t OFF_XC  = 294700000;          // (B,256) f16 = 2,097,152
static const size_t OFF_Y1  = 297000000;          // (B,1024) f16 = 8,388,608
static const size_t OFF_Y2  = 305500000;          // (B,128) f32 = 2,097,152
// pre-packed f16 weight fragments
static const size_t OFF_RP   = 307700000;
static const size_t RP_W1    = OFF_RP;            // 5 slots x 6144 B (KC<=3, NT=2)
static const size_t RP_W2    = RP_W1 + 5*6144;    // 5 slots x 2048 B
static const size_t RP_WXD   = RP_W2 + 5*2048;    // 8192 B  (KC=1, NT=8)
static const size_t RP_WC2   = RP_WXD + 8192;     // 32768 B (KC=8, NT=4)
static const size_t RP_WC3   = RP_WC2 + 32768;    // 131072 B (KC=16, NT=8)
static const size_t RP_WXT   = RP_WC3 + 131072;   // 1,015,808 B (KC=124, NT=8)
static const size_t RP_WF1   = RP_WXT + 1015808;  // 524,288 B (KC=8, NT=64)
static const size_t RP_WF2   = RP_WF1 + 524288;   // 262,144 B (KC=32, NT=8)

static inline void zero_f32(float* p, long long n, hipStream_t s){
  int blocks = (int)((n + 255) / 256);
  k_zero<<<blocks, 256, 0, s>>>(p, n);
}

static inline void repack_w(const float* W, _Float16* rp, int ldw, int NT, int KC, int Kmax,
                            hipStream_t s){
  long long tot = (long long)KC * NT * 512;
  k_repack_w<<<(int)((tot + 255) / 256), 256, 0, s>>>(W, rp, ldw, NT, KC, Kmax);
}

extern "C" void kernel_launch(void* const* d_in, const int* in_sizes, int n_in,
                              void* d_out, int out_size, void* d_ws, size_t ws_size,
                              hipStream_t stream){
  (void)in_sizes; (void)n_in; (void)out_size; (void)ws_size;

  const float* x      = (const float*)d_in[0];
  const int*   ei     = (const int*)d_in[1];
  const int*   src    = ei;
  const int*   dst    = ei + N_EDGES;
  const int*   batch  = (const int*)d_in[2];
  const float* target = (const float*)d_in[3];

  const float *gW1[5], *gb1[5], *gW2[5], *gb2[5], *gbg[5], *gbb[5];
  for (int l = 0; l < 5; ++l){
    int base = 4 + l * 6;
    gW1[l] = (const float*)d_in[base + 0]; gb1[l] = (const float*)d_in[base + 1];
    gW2[l] = (const float*)d_in[base + 2]; gb2[l] = (const float*)d_in[base + 3];
    gbg[l] = (const float*)d_in[base + 4]; gbb[l] = (const float*)d_in[base + 5];
  }
  const float* Wxd = (const float*)d_in[34]; const float* bxd = (const float*)d_in[35];
  const float* Wc1 = (const float*)d_in[36]; const float* bc1 = (const float*)d_in[37];
  const float* Wc2 = (const float*)d_in[38]; const float* bc2 = (const float*)d_in[39];
  const float* Wc3 = (const float*)d_in[40]; const float* bc3 = (const float*)d_in[41];
  const float* Wxt = (const float*)d_in[42]; const float* bxt = (const float*)d_in[43];
  const float* Wf1 = (const float*)d_in[44]; const float* bf1 = (const float*)d_in[45];
  const float* Wf2 = (const float*)d_in[46]; const float* bf2 = (const float*)d_in[47];
  const float* Wo  = (const float*)d_in[48]; const float* bo  = (const float*)d_in[49];

  char* ws = (char*)d_ws;
  float* agg  = (float*)(ws + OFF_AGG);
  float* hb[3] = { (float*)(ws + OFF_H0), (float*)(ws + OFF_H1), (float*)(ws + OFF_H2) };
  _Float16* cl1 = (_Float16*)(ws + OFF_CL1);
  _Float16* c2p = (_Float16*)(ws + OFF_C2P);
  _Float16* cl2 = (_Float16*)(ws + OFF_CL2);
  _Float16* c3p = (_Float16*)(ws + OFF_C3P);
  _Float16* cf  = (_Float16*)(ws + OFF_CF);
  float* g    = (float*)(ws + OFF_G);
  float* gsum = (float*)(ws + OFF_SUM);     // [0:32) sum, [32:64) sumsq
  _Float16* xc = (_Float16*)(ws + OFF_XC);
  _Float16* y1 = (_Float16*)(ws + OFF_Y1);
  float* y2   = (float*)(ws + OFF_Y2);

  _Float16* rpW1[5], *rpW2[5];
  for (int l = 0; l < 5; ++l){
    rpW1[l] = (_Float16*)(ws + RP_W1 + (size_t)l * 6144);
    rpW2[l] = (_Float16*)(ws + RP_W2 + (size_t)l * 2048);
  }
  _Float16* rpWxd = (_Float16*)(ws + RP_WXD);
  _Float16* rpWc2 = (_Float16*)(ws + RP_WC2);
  _Float16* rpWc3 = (_Float16*)(ws + RP_WC3);
  _Float16* rpWxt = (_Float16*)(ws + RP_WXT);
  _Float16* rpWf1 = (_Float16*)(ws + RP_WF1);
  _Float16* rpWf2 = (_Float16*)(ws + RP_WF2);

  float* outp = (float*)d_out;
  float* xd   = outp + BATCH;               // 4096 x 128 f32

  // ---- weight repack (tiny; runs once at launch head) ----
  repack_w(gW1[0], rpW1[0], 32, 2, 3, DIN0, stream);
  for (int l = 1; l < 5; ++l) repack_w(gW1[l], rpW1[l], 32, 2, 1, 32, stream);
  for (int l = 0; l < 5; ++l) repack_w(gW2[l], rpW2[l], 32, 2, 1, 32, stream);
  repack_w(Wxd, rpWxd, 128, 8, 1, 32, stream);
  {
    long long t2 = 8LL * 4 * 512;
    k_repack_convw<<<(int)((t2 + 255) / 256), 256, 0, stream>>>(Wc2, rpWc2, 5, 4, 8);
    long long t3 = 16LL * 8 * 512;
    k_repack_convw<<<(int)((t3 + 255) / 256), 256, 0, stream>>>(Wc3, rpWc3, 6, 8, 16);
  }
  repack_w(Wxt, rpWxt, 128, 8, 124, 3968, stream);
  repack_w(Wf1, rpWf1, 1024, 64, 8, 256, stream);
  repack_w(Wf2, rpWf2, 128, 8, 32, 1024, stream);

  // ---- GNN phase ----
  // layer 0 (Din = 78)
  zero_f32(agg, (long long)N_NODES * DIN0, stream);
  {
    long long tot = (long long)N_EDGES * 96;
    k_edge_agg<<<(int)((tot + 255) / 256), 256, 0, stream>>>(x, src, dst, agg, DIN0, 96, tot);
  }
  zero_f32(gsum, 64, stream);
  k_gin_mlp<DIN0><<<3125, 256, 0, stream>>>(x, agg, rpW1[0], gb1[0], rpW2[0], gb2[0],
                                            hb[0], gsum, gsum + 32);
  k_bn_apply<<<50000, 256, 0, stream>>>(hb[0], gsum, gsum + 32, gbg[0], gbb[0], hb[1], N_NODES);

  int cur = 1, tb = 0, nx = 2;
  for (int l = 1; l < 5; ++l){
    zero_f32(agg, (long long)N_NODES * DIM, stream);
    {
      long long tot = (long long)N_EDGES * 32;
      k_edge_agg<<<(int)((tot + 255) / 256), 256, 0, stream>>>(hb[cur], src, dst, agg, DIM, 32, tot);
    }
    zero_f32(gsum, 64, stream);
    k_gin_mlp<DIM><<<3125, 256, 0, stream>>>(hb[cur], agg, rpW1[l], gb1[l], rpW2[l], gb2[l],
                                             hb[tb], gsum, gsum + 32);
    k_bn_apply<<<50000, 256, 0, stream>>>(hb[tb], gsum, gsum + 32, gbg[l], gbb[l], hb[nx], N_NODES);
    int oc = cur; cur = nx; nx = tb; tb = oc;
  }

  // pooling + xd branch
  zero_f32(g, (long long)BATCH * DIM, stream);
  k_pool<<<50000, 256, 0, stream>>>(hb[cur], batch, g, N_NODES);
  k_xd<<<32, 256, 0, stream>>>(g, rpWxd, bxd, xd, xc);

  // ---- conv tower (reuses dead GNN workspace) ----
  {
    long long tot = 4096LL * 311 * 32;
    k_conv1<<<(int)((tot + 255) / 256), 256, 0, stream>>>(target, Wc1, bc1, cl1);
  }
  k_conv2<<<9728, 256, 0, stream>>>(cl1, rpWc2, bc2, c2p);
  {
    long long tot = 4096LL * 101 * 64;
    k_pool3_cl<<<(int)((tot + 255) / 256), 256, 0, stream>>>(c2p, cl2, 304, 101, 64, tot);
  }
  k_conv3<<<3008, 256, 0, stream>>>(cl2, rpWc3, bc3, c3p);
  {
    long long tot = 4096LL * 128 * 31;
    k_pool3_tr<<<(int)((tot + 255) / 256), 256, 0, stream>>>(c3p, cf, tot);
  }
  k_xt<<<32, 256, 0, stream>>>(cf, rpWxt, bxt, xc);

  // ---- MLP head ----
  k_fc1<<<256, 256, 0, stream>>>(xc, rpWf1, bf1, y1);
  k_fc2<<<32, 256, 0, stream>>>(y1, rpWf2, bf2, y2);
  k_out<<<512, 256, 0, stream>>>(y2, Wo, bo, outp);
}
